// CSS1D_33200097198887
// MI455X (gfx1250) — matrix operand order repaired and checked by reading, not compiled, not run on hardware
//
#include <hip/hip_runtime.h>

// CSS1D (cross selective scan) forward for MI455X / gfx1250.
// Shapes: B=8, L=4096, D_INNER=128, N_STATE=16, DT_RANK=4, K=4 dirs.
#define BATCH 8
#define SEQ   4096
#define DIN   128
#define NST   16
#define RNK   4
#define KDIR  4
#define CPROJ 36          // dt_rank + 2*d_state
#define CPAD  48          // padded to 3 WMMA row-tiles

typedef __attribute__((ext_vector_type(16))) _Float16 v16h;
typedef __attribute__((ext_vector_type(8)))  float    v8f;
typedef __attribute__((ext_vector_type(2)))  float    v2f;
typedef __attribute__((ext_vector_type(4)))  unsigned int v4u_t;
typedef __attribute__((ext_vector_type(8)))  int      v8i_t;
typedef __attribute__((ext_vector_type(4)))  int      v4i_t;

union F16Frag { v16h v; unsigned int u[8]; };

#if defined(__gfx1250__) && __has_builtin(__builtin_amdgcn_tensor_load_to_lds)
#define CSS_HAS_TDM 1
#endif

// Direction permutations: source index in the un-permuted sequence for
// position l of direction k  (matches reference _directions()).
__device__ __forceinline__ int perm_src(int k, int l) {
  switch (k) {
    case 0:  return l;
    case 1:  return SEQ - 1 - l;
    case 2:  return (l < SEQ / 2) ? (2 * l) : (2 * (SEQ - 1 - l) + 1);
    default: return (l < SEQ / 2) ? (2 * l + 1) : (2 * (SEQ - 1 - l));
  }
}

__device__ __forceinline__ float silu_f(float z) { return z / (1.f + __expf(-z)); }

// ---------------------------------------------------------------------------
// Kernel 1: embed = depthwise conv3(x * in_w) + bias, SiLU.
// xp  : fp32 (B,L,D)  -- feeds the scan (u) and the D-residual (exact fp32)
// xch : fp16 (B,L,D)  -- feeds the WMMA projection only
// ---------------------------------------------------------------------------
__global__ void css_embed(const float* __restrict__ x, const float* __restrict__ xc,
                          const float* __restrict__ in_w, const float* __restrict__ in_cw,
                          const float* __restrict__ conv_w, const float* __restrict__ conv_b,
                          float* __restrict__ xp, _Float16* __restrict__ xch) {
  const int idx = blockIdx.x * blockDim.x + threadIdx.x;      // B*L*D
  const int d = idx & (DIN - 1);
  const int l = (idx >> 7) & (SEQ - 1);
  const int b = idx >> 19;
  const float w0 = conv_w[d * 3 + 0], w1 = conv_w[d * 3 + 1], w2 = conv_w[d * 3 + 2];
  const float cb = conv_b[d];
  const int bo = b * SEQ;
  float am = (l > 0)       ? x[bo + l - 1] : 0.f;
  float a0 = x[bo + l];
  float ap = (l < SEQ - 1) ? x[bo + l + 1] : 0.f;
  float z = in_w[d] * (w0 * am + w1 * a0 + w2 * ap) + cb;
  xp[(size_t)(bo + l) * DIN + d] = silu_f(z);
  float cm = (l > 0)       ? xc[bo + l - 1] : 0.f;
  float c0 = xc[bo + l];
  float cp = (l < SEQ - 1) ? xc[bo + l + 1] : 0.f;
  float zc = in_cw[d] * (w0 * cm + w1 * c0 + w2 * cp) + cb;
  xch[(size_t)(bo + l) * DIN + d] = (_Float16)silu_f(zc);
}

// ---------------------------------------------------------------------------
// Kernel 2: convert xproj_w (K,36,128) -> fp16 (K,48,128), zero-padded rows.
// ---------------------------------------------------------------------------
__global__ void css_wcvt(const float* __restrict__ xproj_w, _Float16* __restrict__ Wh) {
  const int idx = blockIdx.x * blockDim.x + threadIdx.x;      // K*CPAD*DIN
  const int d = idx & (DIN - 1);
  const int c = (idx >> 7) % CPAD;
  const int k = idx / (CPAD * DIN);
  const float v = (c < CPROJ) ? xproj_w[(size_t)(k * CPROJ + c) * DIN + d] : 0.f;
  Wh[idx] = (_Float16)v;
}

// ---------------------------------------------------------------------------
// Kernel 3: x_dbl[b,k,c,l] = sum_d W[k,c,d] * xcs[b,k,d,l]  via
// v_wmma_f32_16x16x32_f16.  Three 16x16 (c x l) tiles (48 x 16) per wave; d contracted in
// four K=32 chunks.  A-fragment layout (contiguous k order used by this code):
//   M = lane&15,  K(v,h) = 2*v + 16*h + j   (j packed in dword)
// B-fragment layout (dense 32x16, same contiguous k order as A):
//   N = lane&15,  K(v,h) = 2*v + 16*h + j
// ---------------------------------------------------------------------------
__global__ void __launch_bounds__(256)
css_xdbl_wmma(const _Float16* __restrict__ xch, const _Float16* __restrict__ Wh,
              float* __restrict__ xdbl) {
  __shared__ __align__(16) _Float16 sW[CPAD * DIN];           // 12 KB
  const int tid = threadIdx.x;
  const int bk = blockIdx.x;
  const int b = bk >> 2, k = bk & 3;
  {
    const unsigned int* src = (const unsigned int*)(Wh + (size_t)k * CPAD * DIN);
    unsigned int* dst = (unsigned int*)sW;
    for (int j = tid; j < CPAD * DIN / 2; j += 256) dst[j] = src[j];
  }
  __syncthreads();
  const int lane = tid & 31, wave = tid >> 5;
  const int n = lane & 15, h = lane >> 4;
  const int l0 = (blockIdx.y * 8 + wave) * 16;                // 256 tiles total
  const int lsrc = perm_src(k, l0 + n);
  const _Float16* xrow = xch + (size_t)(b * SEQ + lsrc) * DIN;
  v8f acc0 = {}; v8f acc1 = {}; v8f acc2 = {};
#pragma unroll
  for (int chunk = 0; chunk < 4; ++chunk) {
    const int d0 = chunk * 32;
    F16Frag bf;
#pragma unroll
    for (int v = 0; v < 8; ++v)
      bf.u[v] = *(const unsigned int*)(xrow + d0 + 2 * v + 16 * h);
    F16Frag a0, a1, a2;
#pragma unroll
    for (int v = 0; v < 8; ++v) {
      const int kA = d0 + 2 * v + 16 * h;
      a0.u[v] = *(const unsigned int*)(sW + (0  + n) * DIN + kA);
      a1.u[v] = *(const unsigned int*)(sW + (16 + n) * DIN + kA);
      a2.u[v] = *(const unsigned int*)(sW + (32 + n) * DIN + kA);
    }
    acc0 = __builtin_amdgcn_wmma_f32_16x16x32_f16(false, a0.v, false, bf.v, (short)0, acc0, false, false);
    acc1 = __builtin_amdgcn_wmma_f32_16x16x32_f16(false, a1.v, false, bf.v, (short)0, acc1, false, false);
    acc2 = __builtin_amdgcn_wmma_f32_16x16x32_f16(false, a2.v, false, bf.v, (short)0, acc2, false, false);
  }
  // C/D layout: row M = r + 8*h, col N = lane&15.
  float* obase = xdbl + (size_t)bk * CPAD * SEQ + l0 + n;
#pragma unroll
  for (int r = 0; r < 8; ++r) {
    const int crow = r + 8 * h;
    obase[(size_t)(0  + crow) * SEQ] = acc0[r];
    obase[(size_t)(16 + crow) * SEQ] = acc1[r];
    obase[(size_t)(32 + crow) * SEQ] = acc2[r];
  }
}

// ---------------------------------------------------------------------------
// Kernel 4: delta[b,k,l,d] = softplus( sum_r dt_w[k,d,r]*dts[b,k,r,l] + dt_b )
// Rank-4 contraction == exactly one V_WMMA_F32_16X16X4_F32 per 16x16 tile
// (full fp32, CDNA5-only shape).  A layout per ISA 32-bit 16x4 table:
//   M = lane&15, K = v + 2*h.  B assumed mirrored: N = lane&15, K = v + 2*h.
// ---------------------------------------------------------------------------
__global__ void __launch_bounds__(256)
css_delta(const float* __restrict__ xdbl, const float* __restrict__ dt_w,
          const float* __restrict__ dt_b, float* __restrict__ delta) {
  const int tid = threadIdx.x, lane = tid & 31, wave = tid >> 5;
  const int bk = blockIdx.x, k = bk & 3;
  const int n = lane & 15, h = lane >> 4;
  const int wglobal = blockIdx.y * 8 + wave;                  // 512 waves / bk
#pragma unroll
  for (int i = 0; i < 4; ++i) {
    const int tile = wglobal * 4 + i;                         // 2048 tiles
    const int d0 = (tile & 7) * 16;
    const int l0 = (tile >> 3) * 16;
#if defined(__gfx1250__) && __has_builtin(__builtin_amdgcn_wmma_f32_16x16x4_f32)
    v2f a;
    {
      const float* ap = dt_w + (size_t)(k * DIN + d0 + n) * RNK + 2 * h;
      a.x = ap[0]; a.y = ap[1];
    }
    v2f bf;
    {
      const float* bp = xdbl + (size_t)(bk * CPAD + 2 * h) * SEQ + l0 + n;
      bf.x = bp[0]; bf.y = bp[SEQ];
    }
    v8f acc = {};
    acc = __builtin_amdgcn_wmma_f32_16x16x4_f32(false, a, false, bf, (short)0, acc, false, false);
#pragma unroll
    for (int r = 0; r < 8; ++r) {
      const int d = d0 + r + 8 * h;
      const float val = acc[r] + dt_b[k * DIN + d];
      const float sp = (val > 20.f) ? val : log1pf(__expf(val));
      delta[(size_t)(bk * SEQ + l0 + n) * DIN + d] = sp;
    }
#else
    const float r0 = xdbl[(size_t)(bk * CPAD + 0) * SEQ + l0 + n];
    const float r1 = xdbl[(size_t)(bk * CPAD + 1) * SEQ + l0 + n];
    const float r2 = xdbl[(size_t)(bk * CPAD + 2) * SEQ + l0 + n];
    const float r3 = xdbl[(size_t)(bk * CPAD + 3) * SEQ + l0 + n];
#pragma unroll
    for (int r = 0; r < 8; ++r) {
      const int d = d0 + r + 8 * h;
      const float* wp = dt_w + (size_t)(k * DIN + d) * RNK;
      const float val = wp[0] * r0 + wp[1] * r1 + wp[2] * r2 + wp[3] * r3 + dt_b[k * DIN + d];
      const float sp = (val > 20.f) ? val : log1pf(__expf(val));
      delta[(size_t)(bk * SEQ + l0 + n) * DIN + d] = sp;
    }
#endif
  }
}

#ifdef CSS_HAS_TDM
// Issue a TDM DMA: 2D tile (128 contiguous fp32 cols) x (32 rows, row stride
// SEQ elements) from global `gsrc` into LDS byte offset `lds_off`.
// D# packing per CDNA5 ISA ch.8 (groups 2/3 zero -> dims 2..4 unused).
// This toolchain's builtin is the 6-arg flavor:
//   (u32x4 g0, i32x8 g1, i32x4 g2, i32x4 g3, i32x8 extra, i32 cpol)
__device__ __forceinline__ void tdm_load_bc(const float* gsrc, unsigned lds_off) {
  const unsigned long long ga = (unsigned long long)(size_t)gsrc;
  v4u_t g0;
  g0.x = 1u;                                   // count=1, user descriptor
  g0.y = lds_off;                              // lds_addr (bytes)
  g0.z = (unsigned)(ga & 0xffffffffu);         // global_addr[31:0]
  g0.w = (unsigned)(ga >> 32) | (2u << 30);    // global_addr[56:32] | type=2
  v8i_t g1;
  g1[0] = (int)(2u << 16);                     // data_size = 4 bytes
  g1[1] = (int)((SEQ & 0xffff) << 16);         // tensor_dim0 lo16 (=4096)
  g1[2] = (int)(32u << 16);                    // dim0 hi16=0 | tensor_dim1=32
  g1[3] = (int)(128u << 16);                   // dim1 hi16=0 | tile_dim0=128
  g1[4] = 32;                                  // tile_dim1=32, tile_dim2=0
  g1[5] = SEQ;                                 // tensor_dim0_stride lo32
  g1[6] = 0;                                   // stride hi | dim1_stride lo
  g1[7] = 0;
  v4i_t z4 = {0, 0, 0, 0};
  v8i_t z8 = {0, 0, 0, 0, 0, 0, 0, 0};
  __builtin_amdgcn_tensor_load_to_lds(g0, g1, z4, z4, z8, 0);
}
#endif

// ---------------------------------------------------------------------------
// Kernel 5: sequential selective scan.  One block per (b,k), one thread per
// channel d; 16 fp32 states in registers.  B/C (x_dbl rows 4..35) streamed to
// LDS in 128-step chunks by the Tensor Data Mover (double-buffered, TENSORcnt
// synchronized), overlapping the DMA with the serial exp/FMA chain.
// ---------------------------------------------------------------------------
__global__ void __launch_bounds__(128)
css_scan(const float* __restrict__ xp, const float* __restrict__ xdbl,
         const float* __restrict__ delta, const float* __restrict__ A_logs,
         float* __restrict__ ys) {
  __shared__ __align__(16) float sBC[2][32 * 128];            // 2 x 16 KB
  const int d = threadIdx.x;
  const int bk = blockIdx.x, b = bk >> 2, k = bk & 3;
  float An[NST];
#pragma unroll
  for (int nn = 0; nn < NST; ++nn)
    An[nn] = -__expf(A_logs[(size_t)(k * DIN + d) * NST + nn]);
  float hst[NST];
#pragma unroll
  for (int nn = 0; nn < NST; ++nn) hst[nn] = 0.f;
  const float* bc   = xdbl  + (size_t)(bk * CPAD + RNK) * SEQ;   // rows 4..35
  const float* drow = delta + (size_t)bk * SEQ * DIN + d;
  float*       yrow = ys    + (size_t)bk * SEQ * DIN + d;
#ifdef CSS_HAS_TDM
  const unsigned lds0 = (unsigned)(size_t)(&sBC[0][0]);
  const unsigned lds1 = (unsigned)(size_t)(&sBC[1][0]);
  const bool issuer = (threadIdx.x >> 5) == 0;                // wave 0 only
  if (issuer) tdm_load_bc(bc, lds0);
  int ib = 0;
  for (int t0 = 0; t0 < SEQ; t0 += 128, ib ^= 1) {
    if (issuer) {
      if (t0 + 128 < SEQ) {
        tdm_load_bc(bc + t0 + 128, ib ? lds0 : lds1);         // prefetch next
        __builtin_amdgcn_s_wait_tensorcnt(1);                 // current ready
      } else {
        __builtin_amdgcn_s_wait_tensorcnt(0);
      }
    }
    __syncthreads();                                          // publish chunk
    const float* buf = sBC[ib];
    if (t0 + 128 < SEQ) __builtin_prefetch(drow + (size_t)(t0 + 128) * DIN, 0, 0);
    for (int tr = 0; tr < 128; ++tr) {
      const int t = t0 + tr;
      const float u  = xp[(size_t)(b * SEQ + perm_src(k, t)) * DIN + d];
      const float dt = drow[(size_t)t * DIN];
      const float du = dt * u;
      float y = 0.f;
#pragma unroll
      for (int nn = 0; nn < NST; ++nn) {
        const float dA = __expf(dt * An[nn]);
        hst[nn] = fmaf(dA, hst[nn], du * buf[nn * 128 + tr]);
        y = fmaf(hst[nn], buf[(16 + nn) * 128 + tr], y);
      }
      yrow[(size_t)t * DIN] = y;
    }
    __syncthreads();                                          // chunk consumed
  }
#else
  for (int t0 = 0; t0 < SEQ; t0 += 128) {
    __syncthreads();
#pragma unroll
    for (int i = 0; i < 32; ++i)
      sBC[0][i * 128 + d] = bc[(size_t)i * SEQ + t0 + d];
    if (t0 + 128 < SEQ) __builtin_prefetch(bc + t0 + 128, 0, 0);
    __syncthreads();
    for (int tr = 0; tr < 128; ++tr) {
      const int t = t0 + tr;
      const float u  = xp[(size_t)(b * SEQ + perm_src(k, t)) * DIN + d];
      const float dt = drow[(size_t)t * DIN];
      const float du = dt * u;
      float y = 0.f;
#pragma unroll
      for (int nn = 0; nn < NST; ++nn) {
        const float dA = __expf(dt * An[nn]);
        hst[nn] = fmaf(dA, hst[nn], du * sBC[0][nn * 128 + tr]);
        y = fmaf(hst[nn], sBC[0][(16 + nn) * 128 + tr], y);
      }
      yrow[(size_t)t * DIN] = y;
    }
  }
#endif
}

// ---------------------------------------------------------------------------
// Kernel 6: y += Ds*xs, mean over K, LayerNorm over D, output projection.
// One wave per (b,l); lane owns d = lane + 32*i.
// ---------------------------------------------------------------------------
__global__ void __launch_bounds__(256)
css_finalize(const float* __restrict__ ys, const float* __restrict__ xp,
             const float* __restrict__ Ds, const float* __restrict__ ln_g,
             const float* __restrict__ ln_b, const float* __restrict__ out_w,
             float* __restrict__ out) {
  const int lane = threadIdx.x & 31, wave = threadIdx.x >> 5;
  const int gid = blockIdx.x * 8 + wave;
  const int b = gid >> 12, l = gid & (SEQ - 1);
  float acc[4] = {0.f, 0.f, 0.f, 0.f};
#pragma unroll
  for (int k = 0; k < KDIR; ++k) {
    const float* yr = ys + (size_t)((b * KDIR + k) * SEQ + l) * DIN;
    const float* xr = xp + (size_t)(b * SEQ + perm_src(k, l)) * DIN;
#pragma unroll
    for (int i = 0; i < 4; ++i) {
      const int dd = lane + 32 * i;
      acc[i] += yr[dd] + Ds[k * DIN + dd] * xr[dd];
    }
  }
  float s = 0.f, s2 = 0.f;
#pragma unroll
  for (int i = 0; i < 4; ++i) { acc[i] *= 0.25f; s += acc[i]; s2 += acc[i] * acc[i]; }
#pragma unroll
  for (int off = 16; off; off >>= 1) {
    s  += __shfl_xor(s, off, 32);
    s2 += __shfl_xor(s2, off, 32);
  }
  const float mu = s * (1.f / DIN);
  const float var = s2 * (1.f / DIN) - mu * mu;
  const float rstd = rsqrtf(var + 1e-5f);
  float o = 0.f;
#pragma unroll
  for (int i = 0; i < 4; ++i) {
    const int dd = lane + 32 * i;
    o += ((acc[i] - mu) * rstd * ln_g[dd] + ln_b[dd]) * out_w[dd];
  }
#pragma unroll
  for (int off = 16; off; off >>= 1) o += __shfl_xor(o, off, 32);
  if (lane == 0) out[(size_t)b * SEQ + l] = o;
}

// ---------------------------------------------------------------------------
extern "C" void kernel_launch(void* const* d_in, const int* in_sizes, int n_in,
                              void* d_out, int out_size, void* d_ws, size_t ws_size,
                              hipStream_t stream) {
  const float* x       = (const float*)d_in[0];
  const float* x_cross = (const float*)d_in[1];
  const float* in_w    = (const float*)d_in[2];
  const float* in_cw   = (const float*)d_in[3];
  const float* conv_w  = (const float*)d_in[4];
  const float* conv_b  = (const float*)d_in[5];
  const float* xproj_w = (const float*)d_in[6];
  const float* dt_w    = (const float*)d_in[7];
  const float* dt_b    = (const float*)d_in[8];
  const float* A_logs  = (const float*)d_in[9];
  const float* Ds      = (const float*)d_in[10];
  const float* ln_g    = (const float*)d_in[11];
  const float* ln_b    = (const float*)d_in[12];
  const float* out_w   = (const float*)d_in[13];
  float* out = (float*)d_out;

  // Workspace layout (~176 MB total).
  char* p = (char*)d_ws;
  float*    xp   = (float*)p;    p += (size_t)BATCH * SEQ * DIN * sizeof(float);
  _Float16* xch  = (_Float16*)p; p += (size_t)BATCH * SEQ * DIN * sizeof(_Float16);
  _Float16* Wh   = (_Float16*)p; p += (size_t)KDIR * CPAD * DIN * sizeof(_Float16);
  float*    xdbl = (float*)p;    p += (size_t)BATCH * KDIR * CPAD * SEQ * sizeof(float);
  float*    dlt  = (float*)p;    p += (size_t)BATCH * KDIR * SEQ * DIN * sizeof(float);
  float*    ysb  = (float*)p;

  css_embed<<<BATCH * SEQ * DIN / 256, 256, 0, stream>>>(x, x_cross, in_w, in_cw,
                                                         conv_w, conv_b, xp, xch);
  css_wcvt<<<KDIR * CPAD * DIN / 256, 256, 0, stream>>>(xproj_w, Wh);
  css_xdbl_wmma<<<dim3(BATCH * KDIR, 32), 256, 0, stream>>>(xch, Wh, xdbl);
  css_delta<<<dim3(BATCH * KDIR, 64), 256, 0, stream>>>(xdbl, dt_w, dt_b, dlt);
  css_scan<<<BATCH * KDIR, 128, 0, stream>>>(xp, xdbl, dlt, A_logs, ysb);
  css_finalize<<<BATCH * SEQ / 8, 256, 0, stream>>>(ysb, xp, Ds, ln_g, ln_b, out_w, out);
}
